// ParagraphDecoder_7052336300095
// MI455X (gfx1250) — matrix operand order repaired and checked by reading, not compiled, not run on hardware
//
#include <hip/hip_runtime.h>
#include <hip/hip_bf16.h>
#include <math.h>

// Problem constants (from reference)
#define B_     16
#define T_     100
#define S_     400
#define V_     50000
#define E_     128
#define H_     256
#define EXTV_  51000
#define NEG_   (-1.0e12f)
#define NTILE_V   3125       // V_/16
#define NTILE_EXT 3188       // ceil(EXTV_/16)
#define TILES_PER_WAVE 4

typedef __attribute__((ext_vector_type(16))) __bf16 v16bf;
typedef __attribute__((ext_vector_type(8)))  float  v8f;

__device__ __forceinline__ unsigned short f2bf(float f) {
  unsigned int u = __float_as_uint(f);
  u += 0x7FFFu + ((u >> 16) & 1u);     // round-to-nearest-even
  return (unsigned short)(u >> 16);
}
__device__ __forceinline__ float sigmoidf_(float x) { return 1.0f / (1.0f + expf(-x)); }

// ---------------------------------------------------------------------------
// One-time: pack W_out [H_][V_] fp32 -> bf16 WMMA B-matrix tile layout.
// Layout (16-bit B 32x16): lane n(0..15) holds N=n, K=kb*32+{0..7,16..23};
// lane 16+n holds N=n, K=kb*32+{8..15,24..31}.
// Flat: wp[((tile*8 + kb)*32 + L)*16 + jj] = W_out[kb*32 + (L>>4)*8 + (jj&7) + (jj>>3)*16][tile*16 + (L&15)]
// ---------------------------------------------------------------------------
__global__ __launch_bounds__(256) void kPackWout(const float* __restrict__ W_out,
                                                 unsigned short* __restrict__ wp) {
  size_t idx = (size_t)blockIdx.x * 256 + threadIdx.x;     // 12,800,000 total
  int jj   = (int)(idx & 15);
  int L    = (int)((idx >> 4) & 31);
  int kb   = (int)((idx >> 9) & 7);
  int tile = (int)(idx >> 12);
  int n = tile * 16 + (L & 15);
  int k = kb * 32 + ((L >> 4) * 8) + (jj & 7) + ((jj >> 3) * 16);
  wp[idx] = f2bf(W_out[(size_t)k * V_ + n]);
}

// One-time: memories = encoder_outputs @ W_enc + b_enc   (fp32, stays in L2)
__global__ __launch_bounds__(256) void kMemories(const float* __restrict__ enc,
                                                 const float* __restrict__ W_enc,
                                                 const float* __restrict__ b_enc,
                                                 float* __restrict__ mem) {
  const int bs = blockIdx.x;           // 0..B_*S_-1
  const int j  = threadIdx.x;          // 0..255
  __shared__ float row[H_];
  row[j] = enc[(size_t)bs * H_ + j];
  __syncthreads();
  float acc = b_enc[j];
  for (int k = 0; k < H_; ++k) acc += row[k] * W_enc[k * H_ + j];
  mem[(size_t)bs * H_ + j] = acc;
}

// One-time: state init (h=init_h, c=init_c, ctx=0)
__global__ __launch_bounds__(256) void kInit(const float* __restrict__ ih,
                                             const float* __restrict__ ic,
                                             float* __restrict__ h, float* __restrict__ c,
                                             float* __restrict__ ctx) {
  const int i = blockIdx.x * H_ + threadIdx.x;
  h[i] = ih[i]; c[i] = ic[i]; ctx[i] = 0.0f;
}

// ---------------------------------------------------------------------------
// Per step: one WG per batch row. Full fp32 recurrence:
//   x = [emb[y],ctx]@W_red+b ; LSTM gates ; energy/softmax/ctx ; s=tanh([h,ctx]@W_cat+b)
// Packs s into the WMMA A-matrix bf16 layout, stashes masked energy, and
// (fused) applies the pointer-copy scatter for step t-1 (out row t-1 and the
// previous energy row are complete by stream order; old energy is snapshotted
// into LDS before this step's energy stage overwrites it).
// A layout (16-bit 16x32): lanes 0-15 (M=lane): halves 0-7 -> K 0-7, 8-15 -> K 16-23;
// lanes 16-31 (M=lane-16): halves 0-7 -> K 8-15, 8-15 -> K 24-31.
// ---------------------------------------------------------------------------
__global__ __launch_bounds__(1024) void kRow(
    int t,
    const int* __restrict__ trg_seq, const float* __restrict__ emb,
    const float* __restrict__ W_red, const float* __restrict__ b_red,
    const float* __restrict__ Wx, const float* __restrict__ Wh,
    const float* __restrict__ b_lstm,
    const float* __restrict__ memories, const int* __restrict__ mask,
    const float* __restrict__ W_cat, const float* __restrict__ b_cat,
    float* __restrict__ h, float* __restrict__ c, float* __restrict__ ctx,
    unsigned short* __restrict__ s_pack, float* __restrict__ energy_ws,
    const int* __restrict__ ext_src, float* __restrict__ out)
{
  const int b   = blockIdx.x;
  const int tid = threadIdx.x;

  __shared__ float sh_emb[E_];
  __shared__ float sh_ctx[H_];
  __shared__ float sh_h[H_];
  __shared__ float sh_c[H_];
  __shared__ float sh_x[E_];
  __shared__ float sh_z[1024];
  __shared__ float sh_hn[H_];
  __shared__ float sh_cn[H_];
  __shared__ float sh_e[S_];
  __shared__ float sh_ctxn[H_];
  __shared__ float sh_red[32];
  __shared__ int   sh_ids[S_];   // step t-1 pointer-copy inputs
  __shared__ float sh_ev[S_];

  const int y = trg_seq[b * T_ + t];
  if (tid < E_) sh_emb[tid] = emb[(size_t)y * E_ + tid];
  if (tid < H_) {
    sh_ctx[tid] = ctx[b * H_ + tid];
    sh_h[tid]   = h[b * H_ + tid];
    sh_c[tid]   = c[b * H_ + tid];
  }
  if (t > 0 && tid < S_) {       // snapshot BEFORE this step's energy overwrite
    sh_ids[tid] = ext_src[b * S_ + tid];
    sh_ev[tid]  = energy_ws[b * S_ + tid];
  }
  __syncthreads();

  // x = [emb, ctx] @ W_red + b_red
  if (tid < E_) {
    float acc = b_red[tid];
    for (int k = 0; k < E_; ++k) acc += sh_emb[k] * W_red[k * E_ + tid];
    for (int k = 0; k < H_; ++k) acc += sh_ctx[k] * W_red[(E_ + k) * E_ + tid];
    sh_x[tid] = acc;
  }
  __syncthreads();

  // gates z = x@Wx + h@Wh + b  (1024 outputs, one per thread)
  {
    float acc = b_lstm[tid];
    for (int k = 0; k < E_; ++k) acc += sh_x[k] * Wx[k * 1024 + tid];
    for (int k = 0; k < H_; ++k) acc += sh_h[k] * Wh[k * 1024 + tid];
    sh_z[tid] = acc;
  }
  __syncthreads();

  // LSTM cell update (keras gate order i,f,g,o)
  if (tid < H_) {
    float zi = sh_z[tid], zf = sh_z[H_ + tid], zg = sh_z[2*H_ + tid], zo = sh_z[3*H_ + tid];
    float cn = sigmoidf_(zf) * sh_c[tid] + sigmoidf_(zi) * tanhf(zg);
    float hn = sigmoidf_(zo) * tanhf(cn);
    sh_cn[tid] = cn; sh_hn[tid] = hn;
    c[b * H_ + tid] = cn; h[b * H_ + tid] = hn;
  }
  __syncthreads();

  // energy (masked)
  if (tid < S_) {
    float e;
    if (mask[b * S_ + tid] == 0) e = NEG_;
    else {
      const float* m = memories + ((size_t)b * S_ + tid) * H_;
      e = 0.0f;
      for (int k = 0; k < H_; ++k) e += sh_hn[k] * m[k];
    }
    sh_e[tid] = e;
    energy_ws[b * S_ + tid] = e;
  }
  __syncthreads();

  // softmax: block max
  float val = (tid < S_) ? sh_e[tid] : -3.0e38f;
  for (int off = 16; off > 0; off >>= 1) val = fmaxf(val, __shfl_down(val, off, 32));
  if ((tid & 31) == 0) sh_red[tid >> 5] = val;
  __syncthreads();
  if (tid < 32) {
    float v2 = sh_red[tid];
    for (int off = 16; off > 0; off >>= 1) v2 = fmaxf(v2, __shfl_down(v2, off, 32));
    if (tid == 0) sh_red[0] = v2;
  }
  __syncthreads();
  const float emax = sh_red[0];
  __syncthreads();

  // softmax: block sum
  float ex = (tid < S_) ? expf(sh_e[tid] - emax) : 0.0f;
  float sv = ex;
  for (int off = 16; off > 0; off >>= 1) sv += __shfl_down(sv, off, 32);
  if ((tid & 31) == 0) sh_red[tid >> 5] = sv;
  __syncthreads();
  if (tid < 32) {
    float v2 = sh_red[tid];
    for (int off = 16; off > 0; off >>= 1) v2 += __shfl_down(v2, off, 32);
    if (tid == 0) sh_red[0] = v2;
  }
  __syncthreads();
  const float inv = 1.0f / sh_red[0];
  if (tid < S_) sh_e[tid] = ex * inv;     // attn
  __syncthreads();

  // ctx = attn @ memories  (4 partial sums of 100 positions each)
  {
    const int j = tid & (H_ - 1);
    const int part = tid >> 8;            // 0..3
    float acc = 0.0f;
    const int s0 = part * (S_ / 4);
    for (int s = s0; s < s0 + (S_ / 4); ++s)
      acc += sh_e[s] * memories[((size_t)b * S_ + s) * H_ + j];
    sh_z[tid] = acc;
  }
  __syncthreads();
  if (tid < H_) {
    float cx = sh_z[tid] + sh_z[H_ + tid] + sh_z[2*H_ + tid] + sh_z[3*H_ + tid];
    sh_ctxn[tid] = cx;
    ctx[b * H_ + tid] = cx;
  }
  __syncthreads();

  // s = tanh([h,ctx] @ W_cat + b_cat)   (4 partial sums of 128 k each)
  {
    const int j = tid & (H_ - 1);
    const int part = tid >> 8;
    float acc = (part == 0) ? b_cat[j] : 0.0f;
    const int k0 = part * 128;
    for (int k = k0; k < k0 + 128; ++k) {
      float cv = (k < H_) ? sh_hn[k] : sh_ctxn[k - H_];
      acc += cv * W_cat[k * H_ + j];
    }
    sh_z[tid] = acc;
  }
  __syncthreads();
  if (tid < H_) {
    float sval = tanhf(sh_z[tid] + sh_z[H_ + tid] + sh_z[2*H_ + tid] + sh_z[3*H_ + tid]);
    // pack s[b][k] -> WMMA A layout (M=b)
    const int k  = tid;
    const int kb = k >> 5;
    const int r  = k & 31;
    const int hi = (r >> 3) & 1;                 // r 0-7:0, 8-15:1, 16-23:0, 24-31:1
    const int jj = (r & 7) | ((r & 16) >> 1);    // half index within lane
    const int L  = b + hi * 16;
    s_pack[(size_t)(kb * 32 + L) * 16 + jj] = f2bf(sval);
  }

  // Fused pointer-copy scatter for step t-1: per ext id, elect winner
  // (max energy, ties -> smallest s); winner with e > NEG/2 adds once.
  if (t > 0 && tid < S_) {
    const int   id = sh_ids[tid];
    const float e  = sh_ev[tid];
    bool winner = (e > NEG_ * 0.5f);
    if (winner) {
      for (int s2 = 0; s2 < S_; ++s2) {
        if (sh_ids[s2] == id) {
          const float e2 = sh_ev[s2];
          if (e2 > e || (e2 == e && s2 < tid)) { winner = false; break; }
        }
      }
    }
    if (winner) out[((size_t)b * T_ + (t - 1)) * EXTV_ + id] += e;
  }
}

// ---------------------------------------------------------------------------
// Per step: logits via bf16 WMMA. Each wave keeps the A-matrix (16x256
// activation, 8 x v16bf = 64 VGPRs) resident and sweeps TILES_PER_WAVE
// consecutive 16-column tiles. The B stream is software-pipelined (double
// buffered) so the load for k-block kb+1 is issued before the WMMA for kb,
// keeping L2 reads in flight under the matrix pipe. OOV tail written as zeros.
// D layout: lane<16 -> M=r, lane>=16 -> M=8+r; N = lane&15.
// ---------------------------------------------------------------------------
__global__ __launch_bounds__(256) void kOut(
    int t, const unsigned short* __restrict__ s_pack,
    const unsigned short* __restrict__ wp,
    const float* __restrict__ b_out, float* __restrict__ out)
{
  const int wave  = threadIdx.x >> 5;
  const int lane  = threadIdx.x & 31;
  const int tile0 = (blockIdx.x * 8 + wave) * TILES_PER_WAVE;
  if (tile0 >= NTILE_EXT) return;                // wave-uniform

  const v16bf* Aall = reinterpret_cast<const v16bf*>(s_pack);
  v16bf a[8];
#pragma unroll
  for (int kb = 0; kb < 8; ++kb) a[kb] = Aall[kb * 32 + lane];

  const int mbase = (lane < 16) ? 0 : 8;

#pragma unroll
  for (int i = 0; i < TILES_PER_WAVE; ++i) {
    const int tile = tile0 + i;
    if (tile >= NTILE_EXT) break;                // wave-uniform
    const int n0 = tile * 16;
    const int nn = n0 + (lane & 15);
    if (n0 >= V_) {                              // OOV region: ext part is zero
      if (nn < EXTV_) {
        for (int r = 0; r < 8; ++r)
          out[((size_t)(mbase + r) * T_ + t) * EXTV_ + nn] = 0.0f;
      }
      continue;                                  // wave-uniform
    }
    const float bias = b_out[nn];
    v8f acc;
#pragma unroll
    for (int r = 0; r < 8; ++r) acc[r] = bias;
    const v16bf* Bt = reinterpret_cast<const v16bf*>(wp + (size_t)tile * 4096);
    v16bf bcur = Bt[lane];                       // prologue: k-block 0
#pragma unroll
    for (int kb = 0; kb < 8; ++kb) {
      v16bf bnext = bcur;
      if (kb < 7) bnext = Bt[(kb + 1) * 32 + lane];  // issue next load first
      acc = __builtin_amdgcn_wmma_f32_16x16x32_bf16(false, a[kb], false, bcur,
                                                    (short)0, acc, false, false);
      bcur = bnext;
    }
#pragma unroll
    for (int r = 0; r < 8; ++r)
      out[((size_t)(mbase + r) * T_ + t) * EXTV_ + nn] = acc[r];
  }
}

// ---------------------------------------------------------------------------
// Standalone pointer-copy scatter (used only for the final step t = T-1).
// ---------------------------------------------------------------------------
__global__ __launch_bounds__(512) void kCopy(
    int t, const int* __restrict__ ext_src,
    const float* __restrict__ energy_ws, float* __restrict__ out)
{
  const int b = blockIdx.x;
  const int tid = threadIdx.x;
  __shared__ int   ids[S_];
  __shared__ float ev[S_];
  if (tid < S_) { ids[tid] = ext_src[b * S_ + tid]; ev[tid] = energy_ws[b * S_ + tid]; }
  __syncthreads();
  if (tid < S_) {
    const int   id = ids[tid];
    const float e  = ev[tid];
    bool winner = (e > NEG_ * 0.5f);
    if (winner) {
      for (int s2 = 0; s2 < S_; ++s2) {
        if (ids[s2] == id) {
          const float e2 = ev[s2];
          if (e2 > e || (e2 == e && s2 < tid)) { winner = false; break; }
        }
      }
    }
    if (winner) out[((size_t)b * T_ + t) * EXTV_ + id] += e;
  }
}

extern "C" void kernel_launch(void* const* d_in, const int* in_sizes, int n_in,
                              void* d_out, int out_size, void* d_ws, size_t ws_size,
                              hipStream_t stream) {
  (void)in_sizes; (void)n_in; (void)out_size; (void)ws_size;
  const int*   trg_seq = (const int*)  d_in[0];
  const int*   ext_src = (const int*)  d_in[1];
  const int*   mask    = (const int*)  d_in[2];
  const float* enc_out = (const float*)d_in[3];
  const float* init_h  = (const float*)d_in[4];
  const float* init_c  = (const float*)d_in[5];
  const float* emb     = (const float*)d_in[6];
  const float* W_enc   = (const float*)d_in[7];
  const float* b_enc   = (const float*)d_in[8];
  const float* W_red   = (const float*)d_in[9];
  const float* b_red   = (const float*)d_in[10];
  const float* Wx      = (const float*)d_in[11];
  const float* Wh      = (const float*)d_in[12];
  const float* b_lstm  = (const float*)d_in[13];
  const float* W_cat   = (const float*)d_in[14];
  const float* b_cat   = (const float*)d_in[15];
  const float* W_out   = (const float*)d_in[16];
  const float* b_out   = (const float*)d_in[17];
  float* out = (float*)d_out;

  // Workspace carve-up (256B aligned slices)
  char* w = (char*)d_ws;
  auto take = [&](size_t bytes) -> char* {
    char* p = w;
    w += (bytes + 255) & ~(size_t)255;
    return p;
  };
  float*          memories = (float*)take((size_t)B_ * S_ * H_ * sizeof(float));          // 6.55 MB
  unsigned short* wp       = (unsigned short*)take((size_t)NTILE_V * 4096 * 2);           // 25.6 MB bf16
  float*          hbuf     = (float*)take((size_t)B_ * H_ * sizeof(float));
  float*          cbuf     = (float*)take((size_t)B_ * H_ * sizeof(float));
  float*          ctxbuf   = (float*)take((size_t)B_ * H_ * sizeof(float));
  unsigned short* s_pack   = (unsigned short*)take((size_t)8 * 32 * 16 * 2);
  float*          energy   = (float*)take((size_t)B_ * S_ * sizeof(float));

  // One-time preprocessing
  kPackWout<<<50000, 256, 0, stream>>>(W_out, wp);               // 12.8M elems exactly
  kMemories<<<B_ * S_, 256, 0, stream>>>(enc_out, W_enc, b_enc, memories);
  kInit<<<B_, 256, 0, stream>>>(init_h, init_c, hbuf, cbuf, ctxbuf);

  // Sequential decode: 2 kernels per step (copy for step t-1 fused into kRow t)
  const int outGrid = (NTILE_EXT + 8 * TILES_PER_WAVE - 1) / (8 * TILES_PER_WAVE); // 100
  for (int t = 0; t < T_; ++t) {
    kRow<<<B_, 1024, 0, stream>>>(t, trg_seq, emb, W_red, b_red, Wx, Wh, b_lstm,
                                  memories, mask, W_cat, b_cat,
                                  hbuf, cbuf, ctxbuf, s_pack, energy, ext_src, out);
    kOut<<<outGrid, 256, 0, stream>>>(t, s_pack, wp, b_out, out);
  }
  // Pointer-copy for the last step
  kCopy<<<B_, 512, 0, stream>>>(T_ - 1, ext_src, energy, out);
}
